// RNFSEncoder_52158082843229
// MI455X (gfx1250) — matrix operand order repaired and checked by reading, not compiled, not run on hardware
//
#include <hip/hip_runtime.h>
#include <hip/hip_bf16.h>

typedef __attribute__((ext_vector_type(16))) __bf16 v16bf;
typedef __attribute__((ext_vector_type(8)))  __bf16 v8bf;
typedef __attribute__((ext_vector_type(8)))  float  v8f;

#define NB      16
#define C_IN    32
#define LSEQ    128
#define DIM     256
#define OUT_C   128
#define NPIECES 20
#define L2      16384      // LSEQ*LSEQ
#define FAN1    66         // 2*(C_IN+1)

__device__ __forceinline__ unsigned short f2bf(float f) {
    unsigned int u = __float_as_uint(f);
    u += 0x7FFFu + ((u >> 16) & 1u);        // round-to-nearest-even
    return (unsigned short)(u >> 16);
}

// ---------------------------------------------------------------------------
// Kernel A: ha'[n,d,i] = b1[d] + sum_c w1[d,c]*xm[n,c,i]
//           hb [n,d,i] =          sum_c w1[d,33+c]*xm[n,c,i]
// ---------------------------------------------------------------------------
__global__ __launch_bounds__(256)
void prep_kernel(const float* __restrict__ x, const float* __restrict__ mask,
                 const float* __restrict__ w1, const float* __restrict__ b1,
                 float* __restrict__ haf, float* __restrict__ hbf) {
    int g = blockIdx.x * 256 + threadIdx.x;       // n*DIM*LSEQ + d*LSEQ + i
    int i = g & (LSEQ - 1);
    int d = (g >> 7) & (DIM - 1);
    int n = g >> 15;
    const float* w1r = w1 + d * FAN1;             // uniform per 128-thread group
    float sa = b1[d], sb = 0.0f;
    #pragma unroll 8
    for (int c = 0; c < C_IN; ++c) {
        float xv = x[((n * C_IN + c) << 7) + i];
        sa = fmaf(w1r[c],      xv, sa);
        sb = fmaf(w1r[33 + c], xv, sb);
    }
    float mv = mask[(n << 7) + i];
    sa = fmaf(w1r[32], mv, sa);
    sb = fmaf(w1r[65], mv, sb);
    haf[g] = sa;
    hbf[g] = sb;
}

// ---------------------------------------------------------------------------
// Kernel A2: w2 f32 -> bf16 (row-major OUT_C x DIM)
// ---------------------------------------------------------------------------
__global__ __launch_bounds__(256)
void cvt_w2_kernel(const float* __restrict__ w2, unsigned short* __restrict__ w2bf) {
    int g = blockIdx.x * 256 + threadIdx.x;
    if (g < OUT_C * DIM) w2bf[g] = f2bf(w2[g]);
}

// ---------------------------------------------------------------------------
// Kernel B: per (n,i): y[n,:,i,:] = (w2 @ relu(ha'[:,i] + hb) + b2) / L2
// 128x256x128 GEMM via v_wmma_f32_16x16x32_bf16, h tile staged in LDS (bf16,
// transposed [j][d], 16B-chunk XOR swizzle -> conflict-free B reads, 64KB).
// nt/mi loops kept rolled to cap VGPR pressure (~180/wave -> 4+ waves/SIMD).
// ---------------------------------------------------------------------------
__device__ __forceinline__ int sw_idx(int j, int chunk) {   // 16B chunk address (in ushorts)
    return (j << 8) + (((chunk ^ (j & 15)) & 31) << 3);
}

__global__ __launch_bounds__(128)
void gemm_kernel(const float* __restrict__ haf, const float* __restrict__ hbf,
                 const unsigned short* __restrict__ w2bf, const float* __restrict__ b2,
                 float* __restrict__ ybuf) {
    __shared__ __align__(16) unsigned short hm[LSEQ * DIM];   // 65536 B, swizzled [j][d]
    const int blk = blockIdx.x;              // n*LSEQ + i
    const int n   = blk >> 7;
    const int i   = blk & (LSEQ - 1);
    const int tid = threadIdx.x;

    // Warm L2/WGP$ for the A-matrix rows this wave will consume later
    // (global_prefetch_b8 path). 128 threads cover all 128 w2 rows.
    __builtin_prefetch(&w2bf[tid * DIM], 0, 3);

    // Build h^T tile: hm[j][d] = bf16(relu(ha'[n,d,i] + hb[n,d,j]))
    for (int idx = tid; idx < DIM * LSEQ; idx += 128) {
        int d = idx >> 7;
        int j = idx & (LSEQ - 1);
        float hv = haf[((n * DIM + d) << 7) + i];     // uniform -> scalar load
        float v  = hv + hbf[((n * DIM + d) << 7) + j];
        v = v > 0.0f ? v : 0.0f;
        hm[sw_idx(j, d >> 3) + (d & 7)] = f2bf(v);
    }
    __syncthreads();

    const int wave = tid >> 5;
    const int lane = tid & 31;
    const int half = lane >> 4;      // 0: lanes 0-15, 1: lanes 16-31
    const int rc   = lane & 15;
    const float invL2 = 1.0f / (float)L2;

    #pragma unroll 1
    for (int mi = 0; mi < 2; ++mi) {
        const int mt     = wave * 2 + mi;
        const int o_base = mt * 16;

        // Preload all K-step A fragments for this M tile (ISA 16-bit A layout:
        // per lane two contiguous 8-elem runs at K = half*8 and half*8+16).
        v16bf afr[8];
        #pragma unroll
        for (int kt = 0; kt < 8; ++kt) {
            const unsigned short* ap = w2bf + (o_base + rc) * DIM + kt * 32 + half * 8;
            v8bf lo = *reinterpret_cast<const v8bf*>(ap);
            v8bf hi = *reinterpret_cast<const v8bf*>(ap + 16);
            #pragma unroll
            for (int t = 0; t < 8; ++t) { afr[kt][t] = lo[t]; afr[kt][8 + t] = hi[t]; }
        }
        float bias[8];
        #pragma unroll
        for (int r = 0; r < 8; ++r) bias[r] = b2[o_base + half * 8 + r];

        #pragma unroll 1
        for (int nt = 0; nt < 8; ++nt) {
            const int j = nt * 16 + rc;
            v8f acc = {};
            #pragma unroll
            for (int kt = 0; kt < 8; ++kt) {
                // B layout: per lane two 8-elem runs at K = half*8 and half*8+16 (two 16B chunks)
                int c0 = kt * 4 + half;
                const v8bf* pb0 = reinterpret_cast<const v8bf*>(&hm[sw_idx(j, c0)]);
                const v8bf* pb1 = reinterpret_cast<const v8bf*>(&hm[sw_idx(j, c0 + 2)]);
                v8bf b0 = *pb0, b1v = *pb1;
                v16bf bfr;
                #pragma unroll
                for (int t = 0; t < 8; ++t) { bfr[t] = b0[t]; bfr[8 + t] = b1v[t]; }
                acc = __builtin_amdgcn_wmma_f32_16x16x32_bf16(
                          false, afr[kt], false, bfr, (short)0, acc, false, false);
            }
            // C/D layout: lane -> N=j, VGPR r -> M = r + 8*half
            #pragma unroll
            for (int r = 0; r < 8; ++r) {
                int o = o_base + half * 8 + r;
                ybuf[((n * OUT_C + o) << 14) | (i << 7) | j] = (acc[r] + bias[r]) * invL2;
            }
        }
    }
}

// ---------------------------------------------------------------------------
// Kernel C: per (n,o): descending bitonic sort of 16384 values in LDS,
// piecewise-linear weighted sum, block reduction.
// ---------------------------------------------------------------------------
__global__ __launch_bounds__(256)
void pool_kernel(const float* __restrict__ ybuf, const float* __restrict__ pool_w,
                 float* __restrict__ out) {
    __shared__ float s[L2];                  // 65536 B
    const int blk = blockIdx.x;              // n*OUT_C + o
    const int o   = blk & (OUT_C - 1);
    const int tid = threadIdx.x;
    const float* yrow = ybuf + ((size_t)blk << 14);

    for (int t = tid; t < L2; t += 256) s[t] = yrow[t];
    __syncthreads();

    // Bitonic sort, descending
    for (int k = 2; k <= L2; k <<= 1) {
        for (int jj = k >> 1; jj > 0; jj >>= 1) {
            for (int t = tid; t < L2; t += 256) {
                int ixj = t ^ jj;
                if (ixj > t) {
                    float a = s[t], b = s[ixj];
                    bool up = ((t & k) == 0);
                    if ((a < b) == up) { s[t] = b; s[ixj] = a; }
                }
            }
            __syncthreads();
        }
    }

    // Weighted sum with piecewise-linear weights
    const float* pwr = pool_w + o * (NPIECES + 1);
    float local = 0.0f;
    for (int t = tid; t < L2; t += 256) {
        float pos = (float)t * (1.0f / (float)(L2 - 1));
        pos = pos > 1.0f ? 1.0f : pos;
        float f = (float)NPIECES * pos;
        int idx = (int)f;
        if (idx > NPIECES) idx = NPIECES;
        float frac = f - (float)idx;
        int idx2 = idx + 1;
        if (idx2 > NPIECES) idx2 = NPIECES;
        float w = (1.0f - frac) * pwr[idx] + frac * pwr[idx2];
        local = fmaf(s[t], w, local);
    }
    __syncthreads();
    s[tid] = local;                          // reuse sort buffer for reduction
    __syncthreads();
    for (int step = 128; step > 0; step >>= 1) {
        if (tid < step) s[tid] += s[tid + step];
        __syncthreads();
    }
    if (tid == 0) out[blk] = s[0];
}

// ---------------------------------------------------------------------------
extern "C" void kernel_launch(void* const* d_in, const int* in_sizes, int n_in,
                              void* d_out, int out_size, void* d_ws, size_t ws_size,
                              hipStream_t stream) {
    const float* x      = (const float*)d_in[0];   // (16,32,128)
    const float* mask   = (const float*)d_in[1];   // (16,128)
    const float* w1     = (const float*)d_in[2];   // (256,66)
    const float* b1     = (const float*)d_in[3];   // (256,)
    const float* w2     = (const float*)d_in[4];   // (128,256)
    const float* b2     = (const float*)d_in[5];   // (128,)
    const float* pool_w = (const float*)d_in[6];   // (128,21)
    float* out = (float*)d_out;                    // (16,128)

    // Workspace carve-up
    float* haf = (float*)d_ws;                                    // 524288 f32
    float* hbf = haf + NB * DIM * LSEQ;                           // 524288 f32
    unsigned short* w2bf = (unsigned short*)(hbf + NB * DIM * LSEQ); // 32768 bf16
    float* ybuf = (float*)(w2bf + OUT_C * DIM);                   // 16*128*16384 f32

    prep_kernel<<<(NB * DIM * LSEQ) / 256, 256, 0, stream>>>(x, mask, w1, b1, haf, hbf);
    cvt_w2_kernel<<<(OUT_C * DIM) / 256, 256, 0, stream>>>(w2, w2bf);
    gemm_kernel<<<NB * LSEQ, 128, 0, stream>>>(haf, hbf, w2bf, b2, ybuf);
    pool_kernel<<<NB * OUT_C, 256, 0, stream>>>(ybuf, pool_w, out);
}
